// _TinyAttention_50568944943714
// MI455X (gfx1250) — matrix operand order repaired and checked by reading, not compiled, not run on hardware
//
#include <hip/hip_runtime.h>
#include <hip/hip_bf16.h>
#include <cstdint>

// ---------------------------------------------------------------------------
// Problem constants (fixed by the reference)
// ---------------------------------------------------------------------------
constexpr int Bc   = 2;
constexpr int Sc   = 2048;
constexpr int Dc   = 4096;
constexpr int Pc   = 1024;
constexpr int Hc   = 32;
constexpr int HKVc = 8;
constexpr int HDc  = 128;
constexpr int Tkc  = Pc + Sc;          // 3072
constexpr int GQA  = Hc / HKVc;        // 4
#define ATT_SCALE 0.08838834764831845f // 128^-0.5

// ---------------------------------------------------------------------------
// CDNA5 WMMA types (wave32; v16bf A/B frags, v8f C/D accumulators)
// ---------------------------------------------------------------------------
typedef __attribute__((ext_vector_type(8)))  __bf16 v8bf;
typedef __attribute__((ext_vector_type(16))) __bf16 v16bf;
typedef __attribute__((ext_vector_type(8)))  float  v8f;

// Native CDNA5 f32->bf16 convert (clang lowers to v_cvt* bf16 ops)
__device__ inline __bf16 f2bf(float f) { return static_cast<__bf16>(f); }

// ---------------------------------------------------------------------------
// CDNA5 async global->LDS DMA (ASYNCcnt-tracked, ISA 10.x / 15.18.3 op 98).
// LDS destination address = low 32 bits of the generic shared pointer
// (ISA 10.2: FLAT/async LDS addressing truncates to addr[31:0], wave-relative).
// ---------------------------------------------------------------------------
__device__ inline unsigned lds_addr_of(const void* p) {
    return (unsigned)(uintptr_t)p;
}
__device__ inline void async_ld_b128(unsigned lds_addr, const void* gptr) {
    unsigned long long ga = (unsigned long long)(uintptr_t)gptr;
    asm volatile("global_load_async_to_lds_b128 %0, %1, off"
                 :: "v"(lds_addr), "v"(ga) : "memory");
}
__device__ inline void wait_async0() {
    asm volatile("s_wait_asynccnt 0" ::: "memory");
}

// A-matrix frag (16x32 bf16): lane L holds row M=L%16; elements 0..15 = K
// 2*kb..2*kb+15 (16 contiguous K), kb = (L/16)*8.
__device__ inline v16bf ld_a16(const __bf16* rowp, int kb) {
    v8bf lo = *(const v8bf*)(rowp + 2 * kb);
    v8bf hi = *(const v8bf*)(rowp + 2 * kb + 8);
    return __builtin_shufflevector(lo, hi, 0,1,2,3,4,5,6,7,8,9,10,11,12,13,14,15);
}
// B-matrix frag (32x16 bf16, column-major in LDS): lane L holds col N=L%16,
// 16 contiguous K values starting at (L/16)*16.
__device__ inline v16bf ld_b16(const __bf16* colp) {
    v8bf lo = *(const v8bf*)(colp);
    v8bf hi = *(const v8bf*)(colp + 8);
    return __builtin_shufflevector(lo, hi, 0,1,2,3,4,5,6,7,8,9,10,11,12,13,14,15);
}

// fp32 -> bf16 16-element tile-line conversion into LDS
__device__ inline void cvt16(__bf16* dst, const float* src) {
#pragma unroll
    for (int i = 0; i < 16; i += 4) {
        float4 f = *(const float4*)(src + i);
        dst[i + 0] = f2bf(f.x); dst[i + 1] = f2bf(f.y);
        dst[i + 2] = f2bf(f.z); dst[i + 3] = f2bf(f.w);
    }
}

__device__ inline void st_out(float*  p, float v) { *p = v; }
__device__ inline void st_out(__bf16* p, float v) { *p = f2bf(v); }

// ---------------------------------------------------------------------------
// GEMM: Y[M,N] = X[M,K] * W[N,K]^T   (nn.Linear convention)
// block = 256 threads (8 waves), tile 128x128, BK = 32.
// Wave w computes rows [w*16, w*16+16) x all 128 cols -> 8 v8f accumulators.
// bf16 X operands are DMA'd into LDS with global_load_async_to_lds_b128;
// fp32 operands are converted through native v_cvt on the way into LDS.
// ---------------------------------------------------------------------------
#define GBM 128
#define GBN 128
#define GBK 32
#define GSTR 40   // padded row stride (bf16) -> conflict-free ds_read_b128

template <typename TIn, typename TOut>
__global__ __launch_bounds__(256)
void gemm_xwT(const TIn* __restrict__ X, const float* __restrict__ W,
              TOut* __restrict__ Y, int M, int N, int K) {
    constexpr bool kAsyncA = __is_same(TIn, __bf16);

    __shared__ alignas(16) __bf16 As[GBM * GSTR];
    __shared__ alignas(16) __bf16 Bs[GBN * GSTR];

    const int m0   = blockIdx.y * GBM;
    const int n0   = blockIdx.x * GBN;
    const int t    = threadIdx.x;
    const int wave = t >> 5;
    const int lane = t & 31;
    const int lm   = lane & 15;
    const int lh   = lane >> 4;

    v8f acc[8];
#pragma unroll
    for (int j = 0; j < 8; ++j)
#pragma unroll
        for (int i = 0; i < 8; ++i) acc[j][i] = 0.0f;

    const int lr = t >> 1;          // 0..127 : tile row this thread loads
    const int lc = (t & 1) * 16;    // 0 / 16 : half-row

    for (int k0 = 0; k0 < K; k0 += GBK) {
        const TIn* asrc = X + (size_t)(m0 + lr) * K + k0 + lc;
        if constexpr (kAsyncA) {
            unsigned ad = lds_addr_of(&As[lr * GSTR + lc]);
            async_ld_b128(ad,      asrc);
            async_ld_b128(ad + 16, asrc + 8);
        } else {
            cvt16(&As[lr * GSTR + lc], asrc);
        }
        cvt16(&Bs[lr * GSTR + lc], W + (size_t)(n0 + lr) * K + k0 + lc);
        if (k0 + GBK < K)  // hint next A tile toward the caches
            __builtin_prefetch(X + (size_t)(m0 + lr) * K + k0 + GBK + lc, 0, 1);
        if constexpr (kAsyncA) wait_async0();
        __syncthreads();

        v16bf a = ld_a16(&As[(wave * 16 + lm) * GSTR], lh * 8);
#pragma unroll
        for (int j = 0; j < 8; ++j) {
            v16bf b = ld_b16(&Bs[(j * 16 + lm) * GSTR + lh * 16]);
            acc[j] = __builtin_amdgcn_wmma_f32_16x16x32_bf16(
                false, a, false, b, (short)0, acc[j], false, false);
        }
        __syncthreads();
    }

    // C/D layout: VGPR g -> row (g + 8*(lane/16)), col = lane%16 (ISA 7.12.2)
#pragma unroll
    for (int j = 0; j < 8; ++j) {
#pragma unroll
        for (int g = 0; g < 8; ++g) {
            int m = m0 + wave * 16 + g + 8 * lh;
            int n = n0 + j * 16 + lm;
            st_out(Y + (size_t)m * N + n, acc[j][g]);
        }
    }
}

// ---------------------------------------------------------------------------
// KV cache concat: k/v fp32 outputs (B,HKV,Tk,HD) = [past ; new] (new is
// transposed from projection layout (B,S,HKV,HD)), plus bf16 mirrors for the
// attention kernel's async tile loads.
// ---------------------------------------------------------------------------
__global__ void concat_kv(const float* __restrict__ kp, const float* __restrict__ vp,
                          const float* __restrict__ kn, const float* __restrict__ vn,
                          float* __restrict__ kout, float* __restrict__ vout,
                          __bf16* __restrict__ kbf, __bf16* __restrict__ vbf,
                          size_t n) {
    size_t i = (size_t)blockIdx.x * blockDim.x + threadIdx.x;
    if (i >= n) return;
    int    hd = (int)(i % HDc);
    size_t r  = i / HDc;
    int    tp = (int)(r % Tkc);
    size_t r2 = r / Tkc;
    int    hk = (int)(r2 % HKVc);
    int    b  = (int)(r2 / HKVc);
    float kv, vv;
    if (tp < Pc) {
        size_t s = (((size_t)b * HKVc + hk) * Pc + tp) * HDc + hd;
        kv = kp[s]; vv = vp[s];
    } else {
        size_t s = (((size_t)b * Sc + (tp - Pc)) * HKVc + hk) * HDc + hd;
        kv = kn[s]; vv = vn[s];
    }
    kout[i] = kv;
    vout[i] = vv;
    kbf[i]  = f2bf(kv);
    vbf[i]  = f2bf(vv);
}

// ---------------------------------------------------------------------------
// Flash attention (online softmax), GQA, causal with offset P.
// block = 128 threads (4 waves). blockIdx = (q-tile of 64 rows, head, batch).
// Wave w owns 16 query rows; 32-key K/V tiles are shared via LDS.
// Q and K tiles arrive via global_load_async_to_lds_b128 (ASYNCcnt); V is
// transposed bf16->LDS by VALU; QK^T and P*V run on v_wmma_f32_16x16x32_bf16.
// ---------------------------------------------------------------------------
#define QSTR 136   // 128 + 8 padding (bf16)
#define VSTR 40    // 32 + 8 padding (bf16)

__global__ __launch_bounds__(128)
void attn_fwd(const __bf16* __restrict__ Q, const __bf16* __restrict__ Kc,
              const __bf16* __restrict__ Vc, __bf16* __restrict__ Ctx) {
    __shared__ alignas(16) __bf16 Qs[64 * QSTR];          // [qrow][hd]
    __shared__ alignas(16) __bf16 Ks[32 * QSTR];          // [key][hd]
    __shared__ alignas(16) __bf16 Vs[HDc * VSTR];         // [hd][key] (transposed)
    __shared__ alignas(16) __bf16 Ps[4 * 16 * VSTR];      // per-wave P tile

    const int t    = threadIdx.x;
    const int wave = t >> 5;
    const int lane = t & 31;
    const int lm   = lane & 15;
    const int lh   = lane >> 4;
    const int b    = blockIdx.z;
    const int h    = blockIdx.y;
    const int hk   = h / GQA;
    const int q0b  = blockIdx.x * 64;
    const int q0w  = q0b + wave * 16;

    // ---- async DMA this wave's 16x128 bf16 Q tile into LDS ----
    {
        int r = lane >> 1;
        int c = (lane & 1) * 64;
        const __bf16* src = Q + (((size_t)b * Sc + q0w + r) * Hc + h) * HDc + c;
        unsigned dst = lds_addr_of(&Qs[(wave * 16 + r) * QSTR + c]);
#pragma unroll
        for (int i = 0; i < 8; ++i)
            async_ld_b128(dst + i * 16, src + i * 8);
    }
    wait_async0();
    __syncthreads();

    // Q A-frags, resident for the whole key loop (4 x K=32 chunks over HD=128)
    v16bf qf[4];
#pragma unroll
    for (int c = 0; c < 4; ++c)
        qf[c] = ld_a16(&Qs[(wave * 16 + lm) * QSTR + c * 32], lh * 8);

    v8f   cacc[8];
    float mrow[8], lrow[8];
#pragma unroll
    for (int j = 0; j < 8; ++j)
#pragma unroll
        for (int i = 0; i < 8; ++i) cacc[j][i] = 0.0f;
#pragma unroll
    for (int g = 0; g < 8; ++g) { mrow[g] = -1e30f; lrow[g] = 0.0f; }

    const size_t kvbase = ((size_t)b * HKVc + hk) * Tkc;
    const int kend = q0b + 64 + Pc < Tkc ? q0b + 64 + Pc : Tkc;  // multiple of 32

    __bf16* Psw = &Ps[wave * 16 * VSTR];

    for (int kb = 0; kb < kend; kb += 32) {
        __syncthreads();  // previous tile fully consumed
        // ---- K tile: async DMA 32 keys x 128 bf16 straight into LDS ----
        {
            int r = t >> 2;
            int c = (t & 3) * 32;
            const __bf16* src = Kc + (kvbase + kb + r) * HDc + c;
            unsigned dst = lds_addr_of(&Ks[r * QSTR + c]);
#pragma unroll
            for (int i = 0; i < 4; ++i)
                async_ld_b128(dst + i * 16, src + i * 8);
        }
        // ---- V tile: bf16 load + transpose to [hd][key] ----
        {
            int r = t >> 2;
            int c = (t & 3) * 32;
            const __bf16* src = Vc + (kvbase + kb + r) * HDc + c;
#pragma unroll
            for (int i = 0; i < 32; i += 8) {
                v8bf v = *(const v8bf*)(src + i);
#pragma unroll
                for (int e = 0; e < 8; ++e)
                    Vs[(c + i + e) * VSTR + r] = v[e];
            }
        }
        wait_async0();
        __syncthreads();

        // ---- scores: two 16x16 tiles (keys kb..kb+15, kb+16..kb+31) ----
        v8f s0, s1;
#pragma unroll
        for (int i = 0; i < 8; ++i) { s0[i] = 0.0f; s1[i] = 0.0f; }
#pragma unroll
        for (int c = 0; c < 4; ++c) {
            v16bf b0 = ld_b16(&Ks[lm * QSTR + c * 32 + lh * 16]);
            s0 = __builtin_amdgcn_wmma_f32_16x16x32_bf16(
                false, qf[c], false, b0, (short)0, s0, false, false);
        }
#pragma unroll
        for (int c = 0; c < 4; ++c) {
            v16bf b1 = ld_b16(&Ks[(16 + lm) * QSTR + c * 32 + lh * 16]);
            s1 = __builtin_amdgcn_wmma_f32_16x16x32_bf16(
                false, qf[c], false, b1, (short)0, s1, false, false);
        }

        // ---- online softmax (rows live at g + 8*lh, cols across 16 lanes) ----
        const int kg0 = kb + lm;
        const int kg1 = kb + 16 + lm;
#pragma unroll
        for (int g = 0; g < 8; ++g) {
            const int qg = q0w + g + 8 * lh;
            float a0 = s0[g] * ATT_SCALE; if (kg0 > qg + Pc) a0 = -1e30f;
            float a1 = s1[g] * ATT_SCALE; if (kg1 > qg + Pc) a1 = -1e30f;
            float mx = fmaxf(a0, a1);
            mx = fmaxf(mx, __shfl_xor(mx, 1));
            mx = fmaxf(mx, __shfl_xor(mx, 2));
            mx = fmaxf(mx, __shfl_xor(mx, 4));
            mx = fmaxf(mx, __shfl_xor(mx, 8));     // within 16-lane half
            float mnew  = fmaxf(mrow[g], mx);
            float alpha = __expf(mrow[g] - mnew);
            float p0 = __expf(a0 - mnew);
            float p1 = __expf(a1 - mnew);
            float rs = p0 + p1;
            rs += __shfl_xor(rs, 1);
            rs += __shfl_xor(rs, 2);
            rs += __shfl_xor(rs, 4);
            rs += __shfl_xor(rs, 8);
            lrow[g] = lrow[g] * alpha + rs;
            mrow[g] = mnew;
#pragma unroll
            for (int j = 0; j < 8; ++j) cacc[j][g] *= alpha;
            // P tile (C layout) -> wave-private LDS for A-layout reload
            Psw[(g + 8 * lh) * VSTR + lm]      = f2bf(p0);
            Psw[(g + 8 * lh) * VSTR + 16 + lm] = f2bf(p1);
        }
        // wave-local LDS RAW: DS pipe is in-order per wave; be explicit anyway
        asm volatile("s_wait_dscnt 0" ::: "memory");

        v16bf pa = ld_a16(&Psw[lm * VSTR], lh * 8);
#pragma unroll
        for (int j = 0; j < 8; ++j) {
            v16bf vb = ld_b16(&Vs[(j * 16 + lm) * VSTR + lh * 16]);
            cacc[j] = __builtin_amdgcn_wmma_f32_16x16x32_bf16(
                false, pa, false, vb, (short)0, cacc[j], false, false);
        }
    }

    // ---- normalize and emit ctx in (B, S, H*HD) layout, bf16 ----
#pragma unroll
    for (int j = 0; j < 8; ++j) {
#pragma unroll
        for (int g = 0; g < 8; ++g) {
            int q = q0w + g + 8 * lh;
            float v = cacc[j][g] / lrow[g];
            Ctx[(((size_t)b * Sc + q) * Hc + h) * HDc + j * 16 + lm] = f2bf(v);
        }
    }
}

// ---------------------------------------------------------------------------
// Launch: q/k/v projections -> concat KV cache (fp32 outputs + bf16 mirrors)
// -> flash attention -> output projection. All matrix math on bf16 WMMA.
// ---------------------------------------------------------------------------
extern "C" void kernel_launch(void* const* d_in, const int* in_sizes, int n_in,
                              void* d_out, int out_size, void* d_ws, size_t ws_size,
                              hipStream_t stream) {
    (void)in_sizes; (void)n_in; (void)out_size; (void)ws_size;

    const float* x      = (const float*)d_in[0];
    const float* k_past = (const float*)d_in[1];
    const float* v_past = (const float*)d_in[2];
    const float* Wq     = (const float*)d_in[3];
    const float* Wk     = (const float*)d_in[4];
    const float* Wv     = (const float*)d_in[5];
    const float* Wo     = (const float*)d_in[6];

    float* out  = (float*)d_out;                         // (B,S,D)
    float* kout = out + (size_t)Bc * Sc * Dc;            // (B,HKV,Tk,HD)
    float* vout = kout + (size_t)Bc * HKVc * Tkc * HDc;  // (B,HKV,Tk,HD)

    char* ws = (char*)d_ws;
    __bf16* q_ws   = (__bf16*)(ws);                         // 32 MB (B,S,H,HD) bf16
    __bf16* ctx_ws = (__bf16*)(ws + (size_t)(32u << 20));   // 32 MB (B,S,H,HD) bf16
    float*  kn_ws  = (float*)(ws + (size_t)(64u << 20));    // 16 MB (B,S,HKV,HD) f32
    float*  vn_ws  = (float*)(ws + (size_t)(80u << 20));    // 16 MB
    __bf16* kbf_ws = (__bf16*)(ws + (size_t)(96u << 20));   // 12 MB (B,HKV,Tk,HD) bf16
    __bf16* vbf_ws = (__bf16*)(ws + (size_t)(109u << 20));  // 12 MB

    const int M = Bc * Sc;  // 4096

    // 1) projections
    gemm_xwT<float, __bf16><<<dim3(Dc / GBN, M / GBM), 256, 0, stream>>>(
        x, Wq, q_ws, M, Dc, Dc);
    gemm_xwT<float, float><<<dim3((HKVc * HDc) / GBN, M / GBM), 256, 0, stream>>>(
        x, Wk, kn_ws, M, HKVc * HDc, Dc);
    gemm_xwT<float, float><<<dim3((HKVc * HDc) / GBN, M / GBM), 256, 0, stream>>>(
        x, Wv, vn_ws, M, HKVc * HDc, Dc);

    // 2) KV cache concat (fp32 reference outputs + bf16 mirrors for attention)
    size_t nkv = (size_t)Bc * HKVc * Tkc * HDc;
    concat_kv<<<dim3((unsigned)((nkv + 255) / 256)), 256, 0, stream>>>(
        k_past, v_past, kn_ws, vn_ws, kout, vout, kbf_ws, vbf_ws, nkv);

    // 3) flash attention
    attn_fwd<<<dim3(Sc / 64, Hc, Bc), 128, 0, stream>>>(q_ws, kbf_ws, vbf_ws, ctx_ws);

    // 4) output projection (bf16 X -> async-LDS operand path)
    gemm_xwT<__bf16, float><<<dim3(Dc / GBN, M / GBM), 256, 0, stream>>>(
        ctx_ws, Wo, out, M, Dc, Dc);
}
